// EncoderBlock_13039520710987
// MI455X (gfx1250) — compile-verified, matrix operand order checked by reading, not run on hardware
//
#include <hip/hip_runtime.h>
#include <hip/hip_bf16.h>
#include <cstdint>
#include <cstddef>

// ---------------------------------------------------------------------------
// Types
// ---------------------------------------------------------------------------
typedef __bf16 bf16;
typedef __attribute__((ext_vector_type(16))) __bf16 v16bf;
typedef __attribute__((ext_vector_type(8)))  __bf16 v8bf;
typedef __attribute__((ext_vector_type(8)))  float  v8f;

#define D_MODEL 1024
#define NH      16
#define DKH     64
#define D_FF    4096
#define BATCH   2
#define SEQ     2048
#define ROWS    (BATCH * SEQ)
#define LN_EPS  1e-6f
#define NEGBIG  -1e9f

// ---------------------------------------------------------------------------
// WMMA helpers (CDNA5 wave32, V_WMMA_F32_16X16X32_BF16)
// ---------------------------------------------------------------------------
static __device__ __forceinline__ v8f wmma_bf16(v16bf a, v16bf b, v8f c) {
  return __builtin_amdgcn_wmma_f32_16x16x32_bf16(
      /*neg_a=*/false, a, /*neg_b=*/false, b,
      /*c_mod=*/(short)0, c, /*reuse_a=*/false, /*reuse_b=*/false);
}

// Load a 16x32 bf16 A/B operand fragment from a row-major tile.
// ISA layout (16-bit A 16x32): lanes 0-15 hold row=lane, K in {0-7, 16-23};
// lanes 16-31 hold row=lane-16, K in {8-15, 24-31}.  Two 16B chunks per lane.
static __device__ __forceinline__ v16bf load_ab_frag(const bf16* tile, int ld) {
  const int lane = threadIdx.x & 31;
  const int grp  = lane >> 4;        // which 8-wide K chunk pair
  const int r    = lane & 15;        // row (M for A, N for B)
  const bf16* rp = tile + (size_t)r * ld + grp * 8;
  v8bf c0 = *(const v8bf*)(rp);
  v8bf c1 = *(const v8bf*)(rp + 16);
  v16bf f;
#pragma unroll
  for (int i = 0; i < 8; ++i) { f[i] = c0[i]; f[i + 8] = c1[i]; }
  return f;
}

// Low 32 bits of a flat pointer to __shared__ == wave-relative LDS byte offset
// (flat->LDS aperture mapping truncates to addr[31:0]).
static __device__ __forceinline__ unsigned lds_off(const void* p) {
  return (unsigned)(size_t)p;
}

// Async DMA of one 16B chunk global -> LDS (ASYNCcnt-tracked, no VGPRs).
static __device__ __forceinline__ void async_cp16(unsigned lds, const void* g) {
  asm volatile("global_load_async_to_lds_b128 %0, %1, off"
               :: "v"(lds), "v"(g) : "memory");
}
static __device__ __forceinline__ void wait_async0() {
  asm volatile("s_wait_asynccnt 0x0" ::: "memory");
}

// ---------------------------------------------------------------------------
// f32 -> bf16 conversion (weights), grid-stride
// ---------------------------------------------------------------------------
__global__ __launch_bounds__(256) void cvt_bf16_kernel(
    const float* __restrict__ in, bf16* __restrict__ out, int n) {
  for (int i = blockIdx.x * 256 + threadIdx.x; i < n; i += gridDim.x * 256)
    out[i] = (bf16)in[i];
}

// ---------------------------------------------------------------------------
// LayerNorm (faithful: Bessel ddof=1, divide by (std + eps)), f32 in, bf16 out
// ---------------------------------------------------------------------------
__global__ __launch_bounds__(256) void layernorm_kernel(
    const float* __restrict__ x, const float* __restrict__ alpha,
    const float* __restrict__ beta, bf16* __restrict__ out) {
  __shared__ float red[8];
  const int row = blockIdx.x;
  const int t   = threadIdx.x;
  const float* xr = x + (size_t)row * D_MODEL;
  float v[4];
#pragma unroll
  for (int i = 0; i < 4; ++i) v[i] = xr[t + i * 256];

  float s = v[0] + v[1] + v[2] + v[3];
#pragma unroll
  for (int d = 16; d >= 1; d >>= 1) s += __shfl_xor(s, d, 32);
  if ((t & 31) == 0) red[t >> 5] = s;
  __syncthreads();
  float tot = red[0] + red[1] + red[2] + red[3] +
              red[4] + red[5] + red[6] + red[7];
  const float mean = tot * (1.0f / D_MODEL);

  float sq = 0.f;
#pragma unroll
  for (int i = 0; i < 4; ++i) { float d0 = v[i] - mean; sq += d0 * d0; }
#pragma unroll
  for (int d = 16; d >= 1; d >>= 1) sq += __shfl_xor(sq, d, 32);
  __syncthreads();
  if ((t & 31) == 0) red[t >> 5] = sq;
  __syncthreads();
  float tot2 = red[0] + red[1] + red[2] + red[3] +
               red[4] + red[5] + red[6] + red[7];
  const float stdv  = sqrtf(tot2 * (1.0f / (D_MODEL - 1)));
  const float scale = alpha[0] / (stdv + LN_EPS);
  const float bb    = beta[0];

  bf16* orow = out + (size_t)row * D_MODEL;
#pragma unroll
  for (int i = 0; i < 4; ++i)
    orow[t + i * 256] = (bf16)((v[i] - mean) * scale + bb);
}

// ---------------------------------------------------------------------------
// Tiled bf16 GEMM:  C[M,N] = A[M,K] @ W[N,K]^T  (+bias, +relu, +f32 residual)
// Block tile 128x128, 256 threads (8 waves), wave strip 16x128.
// K-slabs of 32 staged by global_load_async_to_lds_b128 into a ping-pong
// LDS double buffer (one barrier per slab, DMA overlaps the 8 WMMAs).
// mode 0: row-major out (bf16 and/or f32);  mode 1: Q/K [B,H,S,Dk];
// mode 2: V transposed [B,H,Dk,S].
// ---------------------------------------------------------------------------
__global__ __launch_bounds__(256) void gemm_bf16_kernel(
    const bf16* __restrict__ A, const bf16* __restrict__ W,
    const float* __restrict__ bias, const float* __restrict__ resid,
    float* __restrict__ outf, bf16* __restrict__ outb,
    int M, int N, int K, int mode, int relu) {
  __shared__ alignas(16) bf16 As[2][128][40];   // pad 32->40: no bank conflicts
  __shared__ alignas(16) bf16 Bs[2][128][40];

  const int n_blk = blockIdx.x * 128;
  const int m_blk = blockIdx.y * 128;
  const int wave  = threadIdx.x >> 5;
  const int m0    = wave * 16;
  const int t     = threadIdx.x;

  // Per-thread staging coordinates: 512 chunks of 8 bf16 (16B) per matrix.
  int srow[2], scol[2];
#pragma unroll
  for (int i = 0; i < 2; ++i) {
    const int c = t + i * 256;
    srow[i] = c >> 2;
    scol[i] = (c & 3) * 8;
  }

  auto stage = [&](int k0, int buf) {
#pragma unroll
    for (int i = 0; i < 2; ++i) {
      async_cp16(lds_off(&As[buf][srow[i]][scol[i]]),
                 &A[(size_t)(m_blk + srow[i]) * K + (k0 + scol[i])]);
      async_cp16(lds_off(&Bs[buf][srow[i]][scol[i]]),
                 &W[(size_t)(n_blk + srow[i]) * K + (k0 + scol[i])]);
    }
  };

  v8f acc[8] = {};

  stage(0, 0);
  wait_async0();
  __syncthreads();

  int buf = 0;
  for (int k0 = 0; k0 < K; k0 += 32) {
    const bool more = (k0 + 32) < K;
    if (more) stage(k0 + 32, buf ^ 1);   // DMA next slab during compute

    const v16bf a = load_ab_frag(&As[buf][m0][0], 40);
    // Software-pipeline the B fragments: issue next ds_loads before use.
    v16bf bcur = load_ab_frag(&Bs[buf][0][0], 40);
#pragma unroll
    for (int nf = 0; nf < 8; ++nf) {
      v16bf bnxt = bcur;
      if (nf < 7) bnxt = load_ab_frag(&Bs[buf][(nf + 1) * 16][0], 40);
      acc[nf] = wmma_bf16(a, bcur, acc[nf]);
      bcur = bnxt;
    }

    if (more) {
      wait_async0();      // next slab fully landed in LDS
      __syncthreads();    // everyone done reading current slab
      buf ^= 1;
    }
  }

  // Epilogue. C layout: lane = N within 16, M = vgpr + 8*(lane>=16).
  const int lane = threadIdx.x & 31;
  const int half = lane >> 4;
  const int nn   = lane & 15;
#pragma unroll
  for (int nf = 0; nf < 8; ++nf) {
    const int n    = n_blk + nf * 16 + nn;
    const float bv = bias ? bias[n] : 0.f;
#pragma unroll
    for (int r = 0; r < 8; ++r) {
      const int m = m_blk + m0 + half * 8 + r;
      float v = acc[nf][r] + bv;
      if (relu) v = fmaxf(v, 0.f);
      if (resid) v += resid[(size_t)m * N + n];
      if (mode == 0) {
        if (outb) outb[(size_t)m * N + n] = (bf16)v;
        if (outf) outf[(size_t)m * N + n] = v;
      } else {
        const int b_ = m >> 11;          // m / SEQ
        const int q_ = m & (SEQ - 1);
        const int h_ = n >> 6;           // n / DKH
        const int d_ = n & (DKH - 1);
        size_t idx;
        if (mode == 1)  // [B,H,S,Dk]
          idx = (((size_t)(b_ * NH + h_) * SEQ) + q_) * DKH + d_;
        else            // [B,H,Dk,S]  (V transposed for P@V B-operand)
          idx = (((size_t)(b_ * NH + h_) * DKH) + d_) * SEQ + q_;
        outb[idx] = (bf16)v;
      }
    }
  }
}

// ---------------------------------------------------------------------------
// Flash attention: grid (B*H, SEQ/64), 128 threads = 4 waves,
// each wave owns 16 query rows; kv processed 32 at a time.
// Q,K: [B,H,S,64] bf16.  Vt: [B,H,64,S] bf16.  out: [B*S, 1024] bf16.
// ---------------------------------------------------------------------------
__global__ __launch_bounds__(128) void flash_attn_kernel(
    const bf16* __restrict__ Q, const bf16* __restrict__ Kmat,
    const bf16* __restrict__ Vt, const int* __restrict__ mask,
    bf16* __restrict__ out) {
  __shared__ alignas(16) bf16 pt[4][16][40];   // per-wave P tile (16x32, pad)

  const int bh   = blockIdx.x;
  const int b    = bh >> 4;
  const int h    = bh & 15;
  const int wave = threadIdx.x >> 5;
  const int q0   = blockIdx.y * 64 + wave * 16;
  const int lane = threadIdx.x & 31;
  const int half = lane >> 4;
  const int nn   = lane & 15;

  const bf16* Qp = Q    + ((size_t)bh * SEQ + q0) * DKH;
  const bf16* Kp = Kmat + (size_t)bh * SEQ * DKH;
  const bf16* Vp = Vt   + (size_t)bh * DKH * SEQ;
  const int*  mp = mask + b * SEQ;

  const v16bf aq0 = load_ab_frag(Qp,      DKH);   // dk 0..31
  const v16bf aq1 = load_ab_frag(Qp + 32, DKH);   // dk 32..63

  float mi[8], li[8];
#pragma unroll
  for (int r = 0; r < 8; ++r) { mi[r] = -1e30f; li[r] = 0.f; }
  v8f o[4] = {};

  for (int kv = 0; kv < SEQ; kv += 32) {
    // Scores: S = Q @ K^T * 1/sqrt(64); two 16-col groups.
    v8f s0 = {}, s1 = {};
    s0 = wmma_bf16(aq0, load_ab_frag(Kp + (size_t)kv * DKH,             DKH), s0);
    s0 = wmma_bf16(aq1, load_ab_frag(Kp + (size_t)kv * DKH + 32,        DKH), s0);
    s1 = wmma_bf16(aq0, load_ab_frag(Kp + (size_t)(kv + 16) * DKH,      DKH), s1);
    s1 = wmma_bf16(aq1, load_ab_frag(Kp + (size_t)(kv + 16) * DKH + 32, DKH), s1);

    const bool pass0 = mp[kv + nn]      != 0;
    const bool pass1 = mp[kv + 16 + nn] != 0;

    float fac[8];
#pragma unroll
    for (int r = 0; r < 8; ++r) {
      float a = pass0 ? s0[r] * 0.125f : NEGBIG;
      float c = pass1 ? s1[r] * 0.125f : NEGBIG;
      // Row max over the 16 lanes of this half (rows live across lanes).
      float mr = fmaxf(a, c);
      mr = fmaxf(mr, __shfl_xor(mr, 1, 16));
      mr = fmaxf(mr, __shfl_xor(mr, 2, 16));
      mr = fmaxf(mr, __shfl_xor(mr, 4, 16));
      mr = fmaxf(mr, __shfl_xor(mr, 8, 16));
      const float nm = fmaxf(mi[r], mr);
      fac[r] = __expf(mi[r] - nm);
      const float p0 = __expf(a - nm);
      const float p1 = __expf(c - nm);
      float rs = p0 + p1;
      rs += __shfl_xor(rs, 1, 16);
      rs += __shfl_xor(rs, 2, 16);
      rs += __shfl_xor(rs, 4, 16);
      rs += __shfl_xor(rs, 8, 16);
      li[r] = li[r] * fac[r] + rs;
      mi[r] = nm;
      // Stash probabilities: C-layout -> row-major 16x32 tile in LDS.
      pt[wave][half * 8 + r][nn]      = (bf16)p0;
      pt[wave][half * 8 + r][nn + 16] = (bf16)p1;
    }
#pragma unroll
    for (int f = 0; f < 4; ++f)
#pragma unroll
      for (int r = 0; r < 8; ++r) o[f][r] *= fac[r];

    // Re-read P in A-fragment layout (within-wave LDS dependency, no barrier).
    const v16bf ap = load_ab_frag(&pt[wave][0][0], 40);
#pragma unroll
    for (int f = 0; f < 4; ++f)
      o[f] = wmma_bf16(ap, load_ab_frag(Vp + (size_t)(f * 16) * SEQ + kv, SEQ), o[f]);
  }

  // Normalize and write merged-head output row-major [B*S, D_MODEL].
#pragma unroll
  for (int r = 0; r < 8; ++r) {
    const float inv  = 1.f / li[r];
    const size_t row = (size_t)b * SEQ + q0 + half * 8 + r;
#pragma unroll
    for (int f = 0; f < 4; ++f)
      out[row * D_MODEL + h * DKH + f * 16 + nn] = (bf16)(o[f][r] * inv);
  }
}

// ---------------------------------------------------------------------------
// Host-side orchestration
// ---------------------------------------------------------------------------
extern "C" void kernel_launch(void* const* d_in, const int* in_sizes, int n_in,
                              void* d_out, int out_size, void* d_ws, size_t ws_size,
                              hipStream_t stream) {
  const float* x    = (const float*)d_in[0];
  const int*   mask = (const int*)  d_in[1];
  const float* Wq = (const float*)d_in[2];  const float* bq = (const float*)d_in[3];
  const float* Wk = (const float*)d_in[4];  const float* bk = (const float*)d_in[5];
  const float* Wv = (const float*)d_in[6];  const float* bv = (const float*)d_in[7];
  const float* Wo = (const float*)d_in[8];  const float* bo = (const float*)d_in[9];
  const float* W1 = (const float*)d_in[10]; const float* b1 = (const float*)d_in[11];
  const float* W2 = (const float*)d_in[12]; const float* b2 = (const float*)d_in[13];
  const float* alpha1 = (const float*)d_in[14]; const float* bias1 = (const float*)d_in[15];
  const float* alpha2 = (const float*)d_in[16]; const float* bias2 = (const float*)d_in[17];
  float* out = (float*)d_out;

  char* ws = (char*)d_ws;
  size_t off = 0;
  auto alloc = [&](size_t bytes) -> void* {
    void* p = ws + off;
    off = (off + bytes + 255) & ~(size_t)255;
    return p;
  };

  bf16* xn1 = (bf16*)alloc((size_t)ROWS * D_MODEL * 2);
  bf16* Wqb = (bf16*)alloc((size_t)D_MODEL * D_MODEL * 2);
  bf16* Wkb = (bf16*)alloc((size_t)D_MODEL * D_MODEL * 2);
  bf16* Wvb = (bf16*)alloc((size_t)D_MODEL * D_MODEL * 2);
  bf16* Wob = (bf16*)alloc((size_t)D_MODEL * D_MODEL * 2);
  bf16* W1b = (bf16*)alloc((size_t)D_FF * D_MODEL * 2);
  bf16* W2b = (bf16*)alloc((size_t)D_MODEL * D_FF * 2);
  bf16* Qb  = (bf16*)alloc((size_t)ROWS * D_MODEL * 2);
  bf16* Kb  = (bf16*)alloc((size_t)ROWS * D_MODEL * 2);
  bf16* Vtb = (bf16*)alloc((size_t)ROWS * D_MODEL * 2);
  bf16* att = (bf16*)alloc((size_t)ROWS * D_MODEL * 2);
  float* x1 = (float*)alloc((size_t)ROWS * D_MODEL * 4);
  bf16* xn2 = (bf16*)alloc((size_t)ROWS * D_MODEL * 2);
  bf16* hff = (bf16*)alloc((size_t)ROWS * D_FF * 2);

  const int nWsmall = D_MODEL * D_MODEL;
  const int nWbig   = D_FF * D_MODEL;
  cvt_bf16_kernel<<<1024, 256, 0, stream>>>(Wq, Wqb, nWsmall);
  cvt_bf16_kernel<<<1024, 256, 0, stream>>>(Wk, Wkb, nWsmall);
  cvt_bf16_kernel<<<1024, 256, 0, stream>>>(Wv, Wvb, nWsmall);
  cvt_bf16_kernel<<<1024, 256, 0, stream>>>(Wo, Wob, nWsmall);
  cvt_bf16_kernel<<<1024, 256, 0, stream>>>(W1, W1b, nWbig);
  cvt_bf16_kernel<<<1024, 256, 0, stream>>>(W2, W2b, nWbig);

  // LN1
  layernorm_kernel<<<ROWS, 256, 0, stream>>>(x, alpha1, bias1, xn1);

  // QKV projections (scatter into attention-friendly layouts)
  dim3 gD(D_MODEL / 128, ROWS / 128);
  gemm_bf16_kernel<<<gD, 256, 0, stream>>>(xn1, Wqb, bq, nullptr, nullptr, Qb,
                                           ROWS, D_MODEL, D_MODEL, 1, 0);
  gemm_bf16_kernel<<<gD, 256, 0, stream>>>(xn1, Wkb, bk, nullptr, nullptr, Kb,
                                           ROWS, D_MODEL, D_MODEL, 1, 0);
  gemm_bf16_kernel<<<gD, 256, 0, stream>>>(xn1, Wvb, bv, nullptr, nullptr, Vtb,
                                           ROWS, D_MODEL, D_MODEL, 2, 0);

  // Attention
  flash_attn_kernel<<<dim3(BATCH * NH, SEQ / 64), 128, 0, stream>>>(
      Qb, Kb, Vtb, mask, att);

  // Output projection + residual 1 -> x1 (f32)
  gemm_bf16_kernel<<<gD, 256, 0, stream>>>(att, Wob, bo, x, x1, nullptr,
                                           ROWS, D_MODEL, D_MODEL, 0, 0);

  // LN2
  layernorm_kernel<<<ROWS, 256, 0, stream>>>(x1, alpha2, bias2, xn2);

  // FFN
  dim3 gF(D_FF / 128, ROWS / 128);
  gemm_bf16_kernel<<<gF, 256, 0, stream>>>(xn2, W1b, b1, nullptr, nullptr, hff,
                                           ROWS, D_FF, D_MODEL, 0, 1);
  gemm_bf16_kernel<<<gD, 256, 0, stream>>>(hff, W2b, b2, x1, out, nullptr,
                                           ROWS, D_MODEL, D_FF, 0, 0);
}
